// AGTLayer_24927990186121
// MI455X (gfx1250) — matrix operand order repaired and checked by reading, not compiled, not run on hardware
//
#include <hip/hip_runtime.h>

// ---------------------------------------------------------------------------
// AGT layer for MI455X (gfx1250, wave32, WMMA bf16 f32-accum).
//  * sl_i is constant over the softmax axis -> cancels exactly; Wl/al dead.
//  * Flash-style streaming softmax: never materialize the 256MB [B,H,N,N].
//  * fr doubles as value matrix; stored transposed per head so WMMA
//    B-fragments are contiguous 16B loads.
//  * GEMMs use double-buffered GLOBAL_LOAD_ASYNC_TO_LDS (ASYNCcnt) when the
//    toolchain exposes the builtin; falls back to sync staging otherwise.
// ---------------------------------------------------------------------------

#define BB 2
#define NN 2048
#define DD 512
#define HH 8
#define HDIM 64
#define RDIM 16
#define SLOPE 0.01f
#define LN_EPS 1e-5f
#define BETA_C 1.0f

typedef __attribute__((ext_vector_type(16))) __bf16 v16bf;
typedef __attribute__((ext_vector_type(8)))  float  v8f;
typedef unsigned int u32x4 __attribute__((ext_vector_type(4)));
typedef u32x4 u32x4a __attribute__((may_alias));

union Frag {
    v16bf bf;
    u32x4 q[2];
    unsigned short s[16];
};

#if defined(__gfx1250__) && __has_builtin(__builtin_amdgcn_global_load_async_to_lds_b128) && __has_builtin(__builtin_amdgcn_s_wait_asynccnt)
#define ASYNC_OK 1
#else
#define ASYNC_OK 0
#endif

#if ASYNC_OK
// Builtin expects GCC-style int4 vectors with explicit address spaces:
//   param0: v4i __device__* (addrspace(1)), param1: v4i LDS* (addrspace(3))
typedef int v4i_t __attribute__((vector_size(16)));
typedef __attribute__((address_space(1))) v4i_t gas_v4i;
typedef __attribute__((address_space(3))) v4i_t las_v4i;
static __device__ __forceinline__ void async_cp16(const unsigned short* g,
                                                  unsigned short* l) {
    __builtin_amdgcn_global_load_async_to_lds_b128((gas_v4i*)g, (las_v4i*)l,
                                                   0, 0);
}
#endif

static __device__ __forceinline__ unsigned short f2bf(float f) {
    unsigned u = __builtin_bit_cast(unsigned, f);
    u += 0x7fffu + ((u >> 16) & 1u);          // round-to-nearest-even
    return (unsigned short)(u >> 16);
}
static __device__ __forceinline__ unsigned pack2(float a, float b) {
    return (unsigned)f2bf(a) | ((unsigned)f2bf(b) << 16);
}
static __device__ __forceinline__ float leaky(float x) {
    return x >= 0.f ? x : SLOPE * x;
}
static __device__ __forceinline__ v8f wmma_bf16(v16bf a, v16bf b, v8f c) {
    return __builtin_amdgcn_wmma_f32_16x16x32_bf16(false, a, false, b,
                                                   (short)0, c, false, false);
}

// ---------------------------------------------------------------------------
// prep kernels
// ---------------------------------------------------------------------------
__global__ __launch_bounds__(256) void k_cvt_bf16(const float* __restrict__ src,
                                                  unsigned short* __restrict__ dst,
                                                  int n) {
    int i = blockIdx.x * 256 + threadIdx.x;
    int stride = gridDim.x * 256;
    for (; i < n; i += stride) dst[i] = f2bf(src[i]);
}

// WT[j][k] = W[k][j]  (W is [rows=in][cols=out] row-major)
__global__ __launch_bounds__(256) void k_tr_cvt(const float* __restrict__ W,
                                                unsigned short* __restrict__ WT,
                                                int rows, int cols) {
    int i = blockIdx.x * 256 + threadIdx.x;
    if (i >= rows * cols) return;
    int k = i / cols, j = i % cols;
    WT[(size_t)j * rows + k] = f2bf(W[i]);
}

// rk/rq = rh @ Wrs / Wrt  (K=16, tiny -> VALU). Layout: [(b*H+h)*N + n][RDIM]
__global__ __launch_bounds__(256) void k_rproj(const float* __restrict__ rh,
                                               const float* __restrict__ Wrs,
                                               const float* __restrict__ Wrt,
                                               unsigned short* __restrict__ rk,
                                               unsigned short* __restrict__ rq) {
    int i = blockIdx.x * 256 + threadIdx.x;
    if (i >= BB * NN * HH * RDIM) return;
    int c  = i & (HH * RDIM - 1);
    int bn = i >> 7;
    const float* r = rh + (size_t)bn * RDIM;
    float ak = 0.f, aq = 0.f;
#pragma unroll
    for (int k = 0; k < RDIM; ++k) {
        ak += r[k] * Wrs[k * (HH * RDIM) + c];
        aq += r[k] * Wrt[k * (HH * RDIM) + c];
    }
    int b = bn / NN, n = bn % NN, hh = c >> 4, rr = c & 15;
    size_t dst = ((size_t)(b * HH + hh) * NN + n) * RDIM + rr;
    rk[dst] = f2bf(ak);
    rq[dst] = f2bf(aq);
}

// ---------------------------------------------------------------------------
// fr = h @ Wr (WMMA). Workgroup: 64 rows x 64 cols (= one head). Epilogue
// writes V^T (bf16, [bh][hd][n]) and sr[bh][n] = sum_hd leaky(fr)*ar[hd]
// (deterministic shuffle reduction, no FP atomics).
// ---------------------------------------------------------------------------
__global__ __launch_bounds__(256) void k_fr_proj(const unsigned short* __restrict__ h_bf,
                                                 const unsigned short* __restrict__ WrT,
                                                 const float* __restrict__ ar,
                                                 unsigned short* __restrict__ vT,
                                                 float* __restrict__ sr) {
    __shared__ __attribute__((aligned(16))) unsigned short Al[2][64 * 32];
    __shared__ __attribute__((aligned(16))) unsigned short Bl[2][64 * 32];
    __shared__ float s_part[2][64];

    int tid = threadIdx.x;
    int wave = tid >> 5, lane = tid & 31;
    int rl = lane & 15, hi = lane >> 4;
    int bn0 = blockIdx.x * 64;
    int head = blockIdx.y;
    int r0 = (wave & 3) * 16, c0 = (wave >> 2) * 32;

    v8f acc0 = {0.f,0.f,0.f,0.f,0.f,0.f,0.f,0.f};
    v8f acc1 = acc0;

    int la_r = tid >> 2, la_c = (tid & 3) * 8;
    int la_off = la_r * 32 + la_c;
    const unsigned short* gA = h_bf + (size_t)(bn0 + la_r) * DD + la_c;
    const unsigned short* gB = WrT + (size_t)(head * 64 + la_r) * DD + la_c;

#if ASYNC_OK
    async_cp16(gA, &Al[0][la_off]);
    async_cp16(gB, &Bl[0][la_off]);
#endif
    for (int k0 = 0; k0 < DD; k0 += 32) {
#if ASYNC_OK
        int buf = (k0 >> 5) & 1;
        const unsigned short* As = &Al[buf][0];
        const unsigned short* Bs = &Bl[buf][0];
        if (k0 + 32 < DD) {
            async_cp16(gA + k0 + 32, &Al[buf ^ 1][la_off]);
            async_cp16(gB + k0 + 32, &Bl[buf ^ 1][la_off]);
            __builtin_amdgcn_s_wait_asynccnt(2);
        } else {
            __builtin_amdgcn_s_wait_asynccnt(0);
        }
        __syncthreads();
#else
        const unsigned short* As = &Al[0][0];
        const unsigned short* Bs = &Bl[0][0];
        __syncthreads();
        *(u32x4a*)&Al[0][la_off] = *(const u32x4a*)(gA + k0);
        *(u32x4a*)&Bl[0][la_off] = *(const u32x4a*)(gB + k0);
        __syncthreads();
#endif
        Frag a, b0, b1;
        int arow = r0 + rl, kb = hi * 16;
        a.q[0] = *(const u32x4a*)&As[arow * 32 + kb];
        a.q[1] = *(const u32x4a*)&As[arow * 32 + kb + 8];
        int bc0 = c0 + rl, kk = hi * 16;
        b0.q[0] = *(const u32x4a*)&Bs[bc0 * 32 + kk];
        b0.q[1] = *(const u32x4a*)&Bs[bc0 * 32 + kk + 8];
        b1.q[0] = *(const u32x4a*)&Bs[(bc0 + 16) * 32 + kk];
        b1.q[1] = *(const u32x4a*)&Bs[(bc0 + 16) * 32 + kk + 8];
        acc0 = wmma_bf16(a.bf, b0.bf, acc0);
        acc1 = wmma_bf16(a.bf, b1.bf, acc1);
#if ASYNC_OK
        __syncthreads();      // LDS reads done (wmma waited dscnt) before reuse
#endif
    }

    int b = bn0 >> 11, nb = bn0 & (NN - 1);
    int nr = r0 + hi * 8;
    size_t vbase = (size_t)(b * HH + head) * 64;
    {   // V^T: lane owns one hd column, 8 consecutive n values -> one b128
        u32x4 p0 = { pack2(acc0[0], acc0[1]), pack2(acc0[2], acc0[3]),
                     pack2(acc0[4], acc0[5]), pack2(acc0[6], acc0[7]) };
        *(u32x4a*)&vT[(vbase + c0 + rl) * NN + nb + nr] = p0;
        u32x4 p1 = { pack2(acc1[0], acc1[1]), pack2(acc1[2], acc1[3]),
                     pack2(acc1[4], acc1[5]), pack2(acc1[6], acc1[7]) };
        *(u32x4a*)&vT[(vbase + c0 + 16 + rl) * NN + nb + nr] = p1;
    }
    // sr partials: reduce over the 16 lanes (columns) of each half-wave
    float arc0 = ar[c0 + rl], arc1 = ar[c0 + 16 + rl];
    float part[8];
#pragma unroll
    for (int v = 0; v < 8; ++v)
        part[v] = leaky(acc0[v]) * arc0 + leaky(acc1[v]) * arc1;
#pragma unroll
    for (int off = 1; off <= 8; off <<= 1)
#pragma unroll
        for (int v = 0; v < 8; ++v)
            part[v] += __shfl_xor(part[v], off, 32);
    if (rl == 0) {
#pragma unroll
        for (int v = 0; v < 8; ++v)
            s_part[wave >> 2][nr + v] = part[v];
    }
    __syncthreads();
    if (tid < 64)
        sr[(size_t)(b * HH + head) * NN + nb + tid] = s_part[0][tid] + s_part[1][tid];
}

// ---------------------------------------------------------------------------
// Flash attention: wave = 16 query rows, workgroup = 128 rows, stream j in 32s.
// score = beta*(rk@rq^T) + sr_j  (sl_i cancels in softmax; TEMPER=1).
// ---------------------------------------------------------------------------
__global__ __launch_bounds__(256) void k_attn(const unsigned short* __restrict__ rk,
                                              const unsigned short* __restrict__ rq,
                                              const float* __restrict__ sr,
                                              const unsigned short* __restrict__ vT,
                                              unsigned short* __restrict__ hsa) {
    __shared__ __attribute__((aligned(16))) unsigned short Pl[8 * 16 * 32];

    int tid = threadIdx.x, wave = tid >> 5, lane = tid & 31;
    int rl = lane & 15, hi = lane >> 4;
    int bh = blockIdx.y, b = bh >> 3, head = bh & 7;
    int i0 = blockIdx.x * 128 + wave * 16;

    const unsigned short* rkp = rk + ((size_t)bh * NN + i0) * RDIM;
    const unsigned short* rqp = rq + (size_t)bh * NN * RDIM;
    const float* srp = sr + (size_t)bh * NN;
    const unsigned short* vp = vT + (size_t)bh * 64 * NN;
    unsigned short* Pw = &Pl[wave * 512];

    const u32x4 zq = {0u, 0u, 0u, 0u};
    Frag aK;                        // loop-invariant A fragment (K=16, pad->32)
    aK.q[0] = hi ? zq : *(const u32x4a*)&rkp[rl * RDIM];
    aK.q[1] = hi ? zq : *(const u32x4a*)&rkp[rl * RDIM + 8];

    v8f zero = {0.f,0.f,0.f,0.f,0.f,0.f,0.f,0.f};
    v8f Ot[4] = {zero, zero, zero, zero};
    float m[8], l[8];
#pragma unroll
    for (int v = 0; v < 8; ++v) { m[v] = -1e30f; l[v] = 0.f; }

    for (int j0 = 0; j0 < NN; j0 += 32) {
        Frag bq0, bq1;              // B frags: lanes 16-31 cover K=16..31 (=0)
        if (hi == 0) {
            const u32x4a* p0 = (const u32x4a*)&rqp[(j0 + rl) * RDIM];
            bq0.q[0] = p0[0]; bq0.q[1] = p0[1];
            const u32x4a* p1 = (const u32x4a*)&rqp[(j0 + 16 + rl) * RDIM];
            bq1.q[0] = p1[0]; bq1.q[1] = p1[1];
        } else {
            bq0.q[0] = zq; bq0.q[1] = zq; bq1.q[0] = zq; bq1.q[1] = zq;
        }
        v8f s0 = wmma_bf16(aK.bf, bq0.bf, zero);
        v8f s1 = wmma_bf16(aK.bf, bq1.bf, zero);
        float sr0 = srp[j0 + rl], sr1 = srp[j0 + 16 + rl];

        float mn[8];
#pragma unroll
        for (int v = 0; v < 8; ++v) {
            s0[v] = BETA_C * s0[v] + sr0;
            s1[v] = BETA_C * s1[v] + sr1;
            mn[v] = fmaxf(s0[v], s1[v]);
        }
#pragma unroll
        for (int off = 1; off <= 8; off <<= 1)
#pragma unroll
            for (int v = 0; v < 8; ++v)
                mn[v] = fmaxf(mn[v], __shfl_xor(mn[v], off, 32));

        float p0[8], p1[8], rs[8], sc[8];
#pragma unroll
        for (int v = 0; v < 8; ++v) {
            float mnew = fmaxf(m[v], mn[v]);
            sc[v] = __expf(m[v] - mnew);
            m[v] = mnew;
            p0[v] = __expf(s0[v] - mnew);
            p1[v] = __expf(s1[v] - mnew);
            rs[v] = p0[v] + p1[v];
        }
#pragma unroll
        for (int off = 1; off <= 8; off <<= 1)
#pragma unroll
            for (int v = 0; v < 8; ++v)
                rs[v] += __shfl_xor(rs[v], off, 32);
#pragma unroll
        for (int v = 0; v < 8; ++v) l[v] = l[v] * sc[v] + rs[v];
#pragma unroll
        for (int t = 0; t < 4; ++t)
#pragma unroll
            for (int v = 0; v < 8; ++v) Ot[t][v] *= sc[v];

        // C-layout -> A-layout transpose of P through this wave's LDS slab
#pragma unroll
        for (int v = 0; v < 8; ++v) {
            int prow = v + hi * 8;
            Pw[prow * 32 + rl]      = f2bf(p0[v]);
            Pw[prow * 32 + 16 + rl] = f2bf(p1[v]);
        }
        Frag aP;
        aP.q[0] = *(const u32x4a*)&Pw[rl * 32 + hi * 16];
        aP.q[1] = *(const u32x4a*)&Pw[rl * 32 + hi * 16 + 8];
#pragma unroll
        for (int t = 0; t < 4; ++t) {
            Frag bv;                // V^T rows are contiguous in j
            const unsigned short* vb = vp + (size_t)(t * 16 + rl) * NN + j0 + hi * 16;
            bv.q[0] = *(const u32x4a*)vb;
            bv.q[1] = *(const u32x4a*)(vb + 8);
            Ot[t] = wmma_bf16(aP.bf, bv.bf, Ot[t]);
        }
    }

    float inv[8];
#pragma unroll
    for (int v = 0; v < 8; ++v) inv[v] = 1.f / l[v];
#pragma unroll
    for (int t = 0; t < 4; ++t)
#pragma unroll
        for (int v = 0; v < 8; ++v) {
            int n = i0 + v + hi * 8;
            hsa[(size_t)(b * NN + n) * DD + head * HDIM + t * 16 + rl] =
                f2bf(Ot[t][v] * inv[v]);
        }
}

// ---------------------------------------------------------------------------
// x = h + hsa @ Wf   (WMMA GEMM, fused residual)
// ---------------------------------------------------------------------------
__global__ __launch_bounds__(256) void k_out_gemm(const unsigned short* __restrict__ hsa,
                                                  const unsigned short* __restrict__ WfT,
                                                  const float* __restrict__ h,
                                                  float* __restrict__ x) {
    __shared__ __attribute__((aligned(16))) unsigned short Al[2][64 * 32];
    __shared__ __attribute__((aligned(16))) unsigned short Bl[2][64 * 32];

    int tid = threadIdx.x;
    int wave = tid >> 5, lane = tid & 31;
    int rl = lane & 15, hi = lane >> 4;
    int bn0 = blockIdx.x * 64;
    int cb = blockIdx.y * 64;
    int r0 = (wave & 3) * 16, c0 = (wave >> 2) * 32;

    v8f acc0 = {0.f,0.f,0.f,0.f,0.f,0.f,0.f,0.f};
    v8f acc1 = acc0;

    int la_r = tid >> 2, la_c = (tid & 3) * 8;
    int la_off = la_r * 32 + la_c;
    const unsigned short* gA = hsa + (size_t)(bn0 + la_r) * DD + la_c;
    const unsigned short* gB = WfT + (size_t)(cb + la_r) * DD + la_c;

#if ASYNC_OK
    async_cp16(gA, &Al[0][la_off]);
    async_cp16(gB, &Bl[0][la_off]);
#endif
    for (int k0 = 0; k0 < DD; k0 += 32) {
#if ASYNC_OK
        int buf = (k0 >> 5) & 1;
        const unsigned short* As = &Al[buf][0];
        const unsigned short* Bs = &Bl[buf][0];
        if (k0 + 32 < DD) {
            async_cp16(gA + k0 + 32, &Al[buf ^ 1][la_off]);
            async_cp16(gB + k0 + 32, &Bl[buf ^ 1][la_off]);
            __builtin_amdgcn_s_wait_asynccnt(2);
        } else {
            __builtin_amdgcn_s_wait_asynccnt(0);
        }
        __syncthreads();
#else
        const unsigned short* As = &Al[0][0];
        const unsigned short* Bs = &Bl[0][0];
        __syncthreads();
        *(u32x4a*)&Al[0][la_off] = *(const u32x4a*)(gA + k0);
        *(u32x4a*)&Bl[0][la_off] = *(const u32x4a*)(gB + k0);
        __syncthreads();
#endif
        Frag a, b0, b1;
        int arow = r0 + rl, kb = hi * 16;
        a.q[0] = *(const u32x4a*)&As[arow * 32 + kb];
        a.q[1] = *(const u32x4a*)&As[arow * 32 + kb + 8];
        int bc0 = c0 + rl, kk = hi * 16;
        b0.q[0] = *(const u32x4a*)&Bs[bc0 * 32 + kk];
        b0.q[1] = *(const u32x4a*)&Bs[bc0 * 32 + kk + 8];
        b1.q[0] = *(const u32x4a*)&Bs[(bc0 + 16) * 32 + kk];
        b1.q[1] = *(const u32x4a*)&Bs[(bc0 + 16) * 32 + kk + 8];
        acc0 = wmma_bf16(a.bf, b0.bf, acc0);
        acc1 = wmma_bf16(a.bf, b1.bf, acc1);
#if ASYNC_OK
        __syncthreads();
#endif
    }
#pragma unroll
    for (int v = 0; v < 8; ++v) {
        int bn = bn0 + r0 + hi * 8 + v;
        int col0 = cb + c0 + rl, col1 = col0 + 16;
        x[(size_t)bn * DD + col0] = acc0[v] + h[(size_t)bn * DD + col0];
        x[(size_t)bn * DD + col1] = acc1[v] + h[(size_t)bn * DD + col1];
    }
}

// ---------------------------------------------------------------------------
// rowwise LayerNorm (D=512, 256 threads: 2 elems/thread)
// ---------------------------------------------------------------------------
__global__ __launch_bounds__(256) void k_ln(const float* __restrict__ x,
                                            const float* __restrict__ gamma,
                                            const float* __restrict__ beta,
                                            float* __restrict__ out) {
    __shared__ float sh1[8], sh2[8];
    int row = blockIdx.x, tid = threadIdx.x;
    const float* xr = x + (size_t)row * DD;
    float a = xr[tid], b = xr[tid + 256];
    float s = a + b, ss = a * a + b * b;
#pragma unroll
    for (int off = 1; off <= 16; off <<= 1) {
        s  += __shfl_xor(s, off, 32);
        ss += __shfl_xor(ss, off, 32);
    }
    if ((tid & 31) == 0) { sh1[tid >> 5] = s; sh2[tid >> 5] = ss; }
    __syncthreads();
    float S = 0.f, SS = 0.f;
#pragma unroll
    for (int w = 0; w < 8; ++w) { S += sh1[w]; SS += sh2[w]; }
    float mu = S * (1.f / DD);
    float var = SS * (1.f / DD) - mu * mu;
    float rsd = rsqrtf(var + LN_EPS);
    out[(size_t)row * DD + tid]       = (a - mu) * rsd * gamma[tid] + beta[tid];
    out[(size_t)row * DD + tid + 256] = (b - mu) * rsd * gamma[tid + 256] + beta[tid + 256];
}

// ---------------------------------------------------------------------------
extern "C" void kernel_launch(void* const* d_in, const int* in_sizes, int n_in,
                              void* d_out, int out_size, void* d_ws, size_t ws_size,
                              hipStream_t stream) {
    (void)in_sizes; (void)n_in; (void)out_size; (void)ws_size;
    const float* h     = (const float*)d_in[0];
    const float* rh    = (const float*)d_in[1];
    /* d_in[2] = Wl : dead (sl_i cancels in softmax)   */
    const float* Wr    = (const float*)d_in[3];
    /* d_in[4] = al : dead                              */
    const float* ar    = (const float*)d_in[5];
    const float* Wrs   = (const float*)d_in[6];
    const float* Wrt   = (const float*)d_in[7];
    const float* Wf    = (const float*)d_in[8];
    const float* gamma = (const float*)d_in[9];
    const float* beta  = (const float*)d_in[10];
    float* out = (float*)d_out;

    char* ws = (char*)d_ws;
    size_t off = 0;
    auto carve = [&](size_t bytes) -> char* {
        char* p = ws + off;
        off += (bytes + 255) & ~(size_t)255;
        return p;
    };
    unsigned short* h_bf = (unsigned short*)carve((size_t)BB * NN * DD * 2);
    unsigned short* WrT  = (unsigned short*)carve((size_t)DD * DD * 2);
    unsigned short* WfT  = (unsigned short*)carve((size_t)DD * DD * 2);
    unsigned short* rk   = (unsigned short*)carve((size_t)BB * HH * NN * RDIM * 2);
    unsigned short* rq   = (unsigned short*)carve((size_t)BB * HH * NN * RDIM * 2);
    float*          srb  = (float*)carve((size_t)BB * HH * NN * 4);
    unsigned short* vT   = (unsigned short*)carve((size_t)BB * HH * HDIM * NN * 2);
    unsigned short* hsa  = (unsigned short*)carve((size_t)BB * NN * DD * 2);
    float*          xbuf = (float*)carve((size_t)BB * NN * DD * 4);

    k_cvt_bf16<<<2048, 256, 0, stream>>>(h, h_bf, BB * NN * DD);
    k_tr_cvt<<<(DD * DD + 255) / 256, 256, 0, stream>>>(Wr, WrT, DD, DD);
    k_tr_cvt<<<(DD * DD + 255) / 256, 256, 0, stream>>>(Wf, WfT, DD, DD);
    k_rproj<<<(BB * NN * HH * RDIM + 255) / 256, 256, 0, stream>>>(rh, Wrs, Wrt, rk, rq);
    k_fr_proj<<<dim3(BB * NN / 64, HH), 256, 0, stream>>>(h_bf, WrT, ar, vT, srb);
    k_attn<<<dim3(NN / 128, BB * HH), 256, 0, stream>>>(rk, rq, srb, vT, hsa);
    k_out_gemm<<<dim3(BB * NN / 64, DD / 64), 256, 0, stream>>>(hsa, WfT, h, xbuf);
    k_ln<<<BB * NN, 256, 0, stream>>>(xbuf, gamma, beta, out);
}
